// PNANet_6038724018833
// MI455X (gfx1250) — matrix operand order repaired and checked by reading, not compiled, not run on hardware
//
#include <hip/hip_runtime.h>
#include <hip/hip_bf16.h>
#include <math.h>

// ---------------- constants ----------------
#define KNN 7
#define TOW 4
#define F1 8
#define F2 64
#define NG 128
#define NPG 512
#define NTOT (NG * NPG)   // 65536

constexpr double LN6 = 1.791759469228055;
constexpr double LN7 = 1.9459101090932196;
constexpr double LN8 = 2.0794415416798357;
constexpr double AVGD = (100.0 * LN6 + 200.0 * LN7 + 700.0 * LN8) / 1000.0;
constexpr float AMP = (float)(LN8 / AVGD);   // log(k+1)/avg_deg_log
constexpr float ATT = (float)(AVGD / LN8);   // 1/amp

typedef __attribute__((ext_vector_type(16))) _Float16 v16h;
typedef __attribute__((ext_vector_type(8)))  float    v8f;

// ---------------- WMMA helpers (CDNA5 16x16x32 f16 -> f32) ----------------
__device__ __forceinline__ v8f wmma_f16(v16h a, v16h b, v8f c) {
  return __builtin_amdgcn_wmma_f32_16x16x32_f16(false, a, false, b, (short)0, c,
                                                false, false);
}

// A-matrix 16x32 f16 fragment from row-major f16 buffer (rowStride in halfs).
// ISA layout: lane L: M = L&15; VGPR0-3 hold K = (L>>4)*8 + 0..7,
//             VGPR4-7 hold K = 16 + (L>>4)*8 + 0..7.
__device__ __forceinline__ v16h load_a16(const _Float16* base, int rowStride,
                                         int kbase, int lane) {
  const _Float16* p = base + (lane & 15) * rowStride + kbase + ((lane >> 4) * 8);
  v16h a;
#pragma unroll
  for (int i = 0; i < 8; ++i) { a[i] = p[i]; a[i + 8] = p[16 + i]; }
  return a;
}

// B-matrix 32x16 f16 fragment from W^T (rows = output col N, rowStride in halfs).
// Layout: lane L: N = L&15; elems 0-7 hold K = (L>>4)*8 + 0..7, elems 8-15 hold K = 16 + (L>>4)*8 + 0..7.
__device__ __forceinline__ v16h load_b16(const _Float16* WT, int rowStride,
                                         int n0, int kbase, int lane) {
  const _Float16* p = WT + (n0 + (lane & 15)) * rowStride + kbase + ((lane >> 4) * 8);
  v16h b;
#pragma unroll
  for (int i = 0; i < 8; ++i) { b[i] = p[i]; b[i + 8] = p[16 + i]; }
  return b;
}

// ---------------- weight preparation (folds biases & constant scalers) -------
__global__ void k_prep(const float* __restrict__ W_pre1, const float* __restrict__ b_pre1,
                       const float* __restrict__ W_post1, const float* __restrict__ W_pre2,
                       const float* __restrict__ b_pre2, const float* __restrict__ W_post2,
                       const float* __restrict__ W_lin2,
                       float* __restrict__ Wcat1, float* __restrict__ bpre1c,
                       float* __restrict__ Weff1, float* __restrict__ bpre2c,
                       _Float16* __restrict__ WcatT, _Float16* __restrict__ Weff2T,
                       _Float16* __restrict__ Wlin2T) {
  const int tid = threadIdx.x;
  // Wcat1 [8][64]: cols 0..31 = top half (x_i), 32..63 = bottom half (x_j)
  for (int idx = tid; idx < 512; idx += 256) {
    int e = idx >> 6, c = idx & 63;
    int t = (c & 31) >> 3, f = c & 7;
    int row = (c < 32) ? e : (8 + e);
    Wcat1[idx] = W_pre1[t * 128 + row * 8 + f];
  }
  if (tid < 32) { int t = tid >> 3, f = tid & 7; bpre1c[tid] = b_pre1[t * 8 + f]; }
  // Weff1 [4][40][16]: rows 0..7 = x part, rows 8..39 = folded agg part
  for (int idx = tid; idx < 2560; idx += 256) {
    int t = idx / 640, r = (idx / 16) % 40, o = idx & 15;
    const float* Wp = W_post1 + t * 1664;  // [104][16]
    float v;
    if (r < 8) v = Wp[r * 16 + o];
    else {
      int j = r - 8;
      v = Wp[(8 + j) * 16 + o] + AMP * Wp[(40 + j) * 16 + o] + ATT * Wp[(72 + j) * 16 + o];
    }
    Weff1[idx] = v;
  }
  if (tid < 256) bpre2c[tid] = b_pre2[tid];  // [4][64] contiguous == col order
  // WcatT [512][64] f16 : row = output col of [W_top | W_bot], col = input e
  for (int idx = tid; idx < 512 * 64; idx += 256) {
    int c = idx >> 6, e = idx & 63;
    int cc = (c < 256) ? c : c - 256;
    int t = cc >> 6, f = cc & 63;
    int row = (c < 256) ? e : (64 + e);
    WcatT[idx] = (_Float16)W_pre2[t * 8192 + row * 64 + f];
  }
  // Weff2T [4][32][320] f16 : k<64 -> x rows; k>=64 -> folded agg rows
  for (int idx = tid; idx < 4 * 32 * 320; idx += 256) {
    int t = idx / (32 * 320), rem = idx % (32 * 320);
    int n = rem / 320, k = rem % 320;
    const float* Wp = W_post2 + t * 26624;  // [832][32]
    float v;
    if (k < 64) v = Wp[k * 32 + n];
    else {
      int j = k - 64;
      v = Wp[(64 + j) * 32 + n] + AMP * Wp[(320 + j) * 32 + n] + ATT * Wp[(576 + j) * 32 + n];
    }
    Weff2T[idx] = (_Float16)v;
  }
  // Wlin2T [128][128] f16 (transposed)
  for (int idx = tid; idx < 16384; idx += 256) {
    int n = idx >> 7, k = idx & 127;
    Wlin2T[idx] = (_Float16)W_lin2[k * 128 + n];
  }
}

// ---------------- kNN: one block per graph ----------------
__global__ void k_knn(const float* __restrict__ pos, int* __restrict__ nbr) {
  __shared__ float px[NPG], py[NPG], pz[NPG];
  const int g = blockIdx.x, tid = threadIdx.x;
  for (int i = tid; i < NPG; i += 256) {
    const float* p = pos + (size_t)(g * NPG + i) * 3;
    px[i] = p[0]; py[i] = p[1]; pz[i] = p[2];
  }
  __syncthreads();
  for (int tt = 0; tt < 2; ++tt) {
    int tgt = tid + 256 * tt;
    float tx = px[tgt], ty = py[tgt], tz = pz[tgt];
    float bd[KNN]; int bi[KNN];
#pragma unroll
    for (int r = 0; r < KNN; ++r) { bd[r] = 3.0e38f; bi[r] = 0; }
    for (int j = 0; j < NPG; ++j) {
      float dx = px[j] - tx, dy = py[j] - ty, dz = pz[j] - tz;
      float d = dx * dx + dy * dy + dz * dz;
      if (j == tgt) d = 1e10f;  // matches reference's diagonal mask
      if (d < bd[KNN - 1]) {
        float cd = d; int ci = j;
#pragma unroll
        for (int r = 0; r < KNN; ++r) {
          if (cd < bd[r]) {  // strict: ties keep lower index first (top_k order)
            float td = bd[r]; int ti = bi[r];
            bd[r] = cd; bi[r] = ci; cd = td; ci = ti;
          }
        }
      }
    }
    int* out = nbr + (size_t)(g * NPG + tgt) * KNN;
#pragma unroll
    for (int r = 0; r < KNN; ++r) out[r] = g * NPG + bi[r];
  }
}

// ---------------- layer 1 (F=8): fp32 VALU ----------------
// P1[n][0:32]  = x_n * Wtop1 + b_pre   (per-target part of m)
// P1[n][32:64] = x_n * Wbot1           (per-source part of m)
__global__ void k_l1_pre(const float* __restrict__ X, const float* __restrict__ Wcat1,
                         const float* __restrict__ bpre1c, float* __restrict__ P1) {
  __shared__ float W[512];
  __shared__ float bc[32];
  const int tid = threadIdx.x;
  for (int i = tid; i < 512; i += 256) W[i] = Wcat1[i];
  if (tid < 32) bc[tid] = bpre1c[tid];
  __syncthreads();
  int node = blockIdx.x * 4 + (tid >> 6);
  int c = tid & 63;
  const float* xr = X + (size_t)node * 8;
  float acc = (c < 32) ? bc[c] : 0.f;
#pragma unroll
  for (int e = 0; e < 8; ++e) acc += xr[e] * W[e * 64 + c];
  P1[(size_t)node * 64 + c] = acc;
}

// agg + post-MLP (folded scalers) + mixing linear; 32 nodes / block, 128 thr
__global__ void k_l1_agg_post(const float* __restrict__ X, const float* __restrict__ P1,
                              const int* __restrict__ nbr, const float* __restrict__ Weff1,
                              const float* __restrict__ Wlin1_g,
                              const float* __restrict__ bpost1,
                              const float* __restrict__ blin1, float* __restrict__ H1) {
  __shared__ float WE[2560];
  __shared__ float WL[4096];
  __shared__ float PO[32][65];
  const int tid = threadIdx.x;
  for (int i = tid; i < 2560; i += 128) WE[i] = Weff1[i];
  for (int i = tid; i < 4096; i += 128) WL[i] = Wlin1_g[i];
  __syncthreads();
  const int i = tid >> 2;        // local node 0..31
  const int t = tid & 3;         // tower
  const int ng = blockIdx.x * 32 + i;
  float pa[8], s[8], mn[8], mx[8], sq[8];
  const float* par = P1 + (size_t)ng * 64 + t * 8;
#pragma unroll
  for (int f = 0; f < 8; ++f) {
    pa[f] = par[f]; s[f] = 0.f; sq[f] = 0.f; mn[f] = 3.0e38f; mx[f] = -3.0e38f;
  }
  for (int k = 0; k < KNN; ++k) {
    int nb = nbr[(size_t)ng * KNN + k];
    const float* pbr = P1 + (size_t)nb * 64 + 32 + t * 8;
#pragma unroll
    for (int f = 0; f < 8; ++f) {
      float m = pa[f] + pbr[f];
      s[f] += m; sq[f] += m * m;
      mn[f] = fminf(mn[f], m); mx[f] = fmaxf(mx[f], m);
    }
  }
  float mean[8], sd[8];
#pragma unroll
  for (int f = 0; f < 8; ++f) {
    mean[f] = s[f] * (1.0f / 7.0f);
    float var = sq[f] * (1.0f / 7.0f) - mean[f] * mean[f];
    sd[f] = sqrtf(fmaxf(var, 0.f) + 1e-5f);
  }
  const float* xr = X + (size_t)ng * 8;
  const float* W = WE + t * 640;  // [40][16]
#pragma unroll
  for (int o = 0; o < 16; ++o) {
    float acc = bpost1[t * 16 + o];
#pragma unroll
    for (int e = 0; e < 8; ++e) acc += xr[e] * W[e * 16 + o];
#pragma unroll
    for (int f = 0; f < 8; ++f)
      acc += mean[f] * W[(8 + f) * 16 + o] + mn[f] * W[(16 + f) * 16 + o] +
             mx[f] * W[(24 + f) * 16 + o] + sd[f] * W[(32 + f) * 16 + o];
    PO[i][t * 16 + o] = acc;
  }
  __syncthreads();
  // mixing linear: thread (node i, quarter t) computes 16 output channels
#pragma unroll 1
  for (int oo = 0; oo < 16; ++oo) {
    int c = t * 16 + oo;
    float acc = blin1[c];
    for (int k = 0; k < 64; ++k) acc += PO[i][k] * WL[k * 64 + c];
    H1[(size_t)ng * 64 + c] = acc;
  }
}

// ---------------- BatchNorm stats (one block per channel) ----------------
__global__ void k_bnstats(const float* __restrict__ H, int C, int n,
                          float* __restrict__ mean, float* __restrict__ istd) {
  const int c = blockIdx.x;
  __shared__ float ss[256], qq[256];
  float s = 0.f, q = 0.f;
  for (int i = threadIdx.x; i < n; i += 256) {
    float v = H[(size_t)i * C + c];
    s += v; q += v * v;
  }
  ss[threadIdx.x] = s; qq[threadIdx.x] = q;
  __syncthreads();
  for (int off = 128; off > 0; off >>= 1) {
    if (threadIdx.x < off) {
      ss[threadIdx.x] += ss[threadIdx.x + off];
      qq[threadIdx.x] += qq[threadIdx.x + off];
    }
    __syncthreads();
  }
  if (threadIdx.x == 0) {
    float m = ss[0] / (float)n;
    float v = qq[0] / (float)n - m * m;
    mean[c] = m;
    istd[c] = rsqrtf(fmaxf(v, 0.f) + 1e-5f);
  }
}

// BN + ReLU -> f16
__global__ void k_bn_relu_half(const float* __restrict__ H, const float* __restrict__ mean,
                               const float* __restrict__ istd, const float* __restrict__ g,
                               const float* __restrict__ b, _Float16* __restrict__ out,
                               int C, int total) {
  int idx = blockIdx.x * 256 + threadIdx.x;
  if (idx >= total) return;
  int c = idx & (C - 1);
  float v = g[c] * (H[idx] - mean[c]) * istd[c] + b[c];
  out[idx] = (_Float16)fmaxf(v, 0.f);
}

// ---------------- layer 2 pre-GEMM: P2 = X2 [N,64] x Wcat [64,512] (WMMA) ----
__global__ void k2_pre(const _Float16* __restrict__ X2h, const _Float16* __restrict__ WcatT,
                       const float* __restrict__ bpre2c, float* __restrict__ PA2,
                       float* __restrict__ PB2) {
  const int m0 = blockIdx.x * 16;
  const int tid = threadIdx.x;
  const int wave = tid >> 5, lane = tid & 31;
  const _Float16* Arow = X2h + (size_t)m0 * 64;
  v16h a0 = load_a16(Arow, 64, 0, lane);
  v16h a1 = load_a16(Arow, 64, 32, lane);
  for (int nt = 0; nt < 4; ++nt) {
    int n0 = (wave * 4 + nt) * 16;  // 0..496
    v16h b0 = load_b16(WcatT, 64, n0, 0, lane);
    v16h b1 = load_b16(WcatT, 64, n0, 32, lane);
    v8f c = {};
    c = wmma_f16(a0, b0, c);
    c = wmma_f16(a1, b1, c);
    const int col = n0 + (lane & 15);
    const int rbase = m0 + ((lane >> 4) * 8);
    // n0 (and therefore col<256) is wave-uniform: force a scalar branch so the
    // epilogue is two clean store clauses instead of per-element exec masking.
    if (__builtin_amdgcn_readfirstlane(n0) < 256) {
      float bias = bpre2c[col];
      float* dst = PA2 + (size_t)rbase * 256 + col;
#pragma unroll
      for (int v = 0; v < 8; ++v) dst[(size_t)v * 256] = c[v] + bias;
    } else {
      float* dst = PB2 + (size_t)rbase * 256 + (col - 256);
#pragma unroll
      for (int v = 0; v < 8; ++v) dst[(size_t)v * 256] = c[v];
    }
  }
}

// ---------------- layer 2 fused: gather+stats -> post-GEMM -> lin-GEMM -------
#define ASTR 1090   // odd-dword LDS row stride (1088 cols + pad) -> no bank conflicts
__global__ void k2_agg_post(const float* __restrict__ PA2, const float* __restrict__ PB2,
                            const _Float16* __restrict__ X2h, const int* __restrict__ nbr,
                            const _Float16* __restrict__ Weff2T,
                            const _Float16* __restrict__ Wlin2T,
                            const float* __restrict__ bpost2,
                            const float* __restrict__ blin2, float* __restrict__ H2) {
  __shared__ _Float16 A[16 * ASTR];   // [16 nodes][x(64) | per-tower mean|min|max|std (4*256)]
  __shared__ _Float16 A2[16 * 130];   // post-MLP result, staged f16 for lin GEMM
  __shared__ int nb[16 * KNN];
  const int m0 = blockIdx.x * 16;
  const int tid = threadIdx.x;
  if (tid < 16 * KNN) nb[tid] = nbr[(size_t)m0 * KNN + tid];
  for (int idx = tid; idx < 16 * 64; idx += 256) {
    int i = idx >> 6, c = idx & 63;
    A[i * ASTR + c] = X2h[(size_t)(m0 + i) * 64 + c];
  }
  __syncthreads();
  // ---- phase 1: f32 stats over 7 neighbors (thread = node x 16-feat chunk) ----
  {
    const int i = tid >> 4;
    const int f0 = (tid & 15) * 16;
    float pa[16], s[16], mn[16], mx[16], sq[16];
    const float* par = PA2 + (size_t)(m0 + i) * 256 + f0;
#pragma unroll
    for (int j = 0; j < 16; ++j) {
      pa[j] = par[j]; s[j] = 0.f; sq[j] = 0.f; mn[j] = 3.0e38f; mx[j] = -3.0e38f;
    }
    for (int k = 0; k < KNN; ++k) {
      const float* pbr = PB2 + (size_t)nb[i * KNN + k] * 256 + f0;
#pragma unroll
      for (int j = 0; j < 16; ++j) {
        float m = pa[j] + pbr[j];
        s[j] += m; sq[j] += m * m;
        mn[j] = fminf(mn[j], m); mx[j] = fmaxf(mx[j], m);
      }
    }
#pragma unroll
    for (int j = 0; j < 16; ++j) {
      int fg = f0 + j, t = fg >> 6, fl = fg & 63;
      float mean = s[j] * (1.0f / 7.0f);
      float var = sq[j] * (1.0f / 7.0f) - mean * mean;
      float sd = sqrtf(fmaxf(var, 0.f) + 1e-5f);
      _Float16* ap = &A[i * ASTR + 64 + t * 256 + fl];
      ap[0] = (_Float16)mean;
      ap[64] = (_Float16)mn[j];
      ap[128] = (_Float16)mx[j];
      ap[192] = (_Float16)sd;
    }
  }
  __syncthreads();
  const int wave = tid >> 5, lane = tid & 31;
  // ---- phase 2: post-MLP per tower, K=320 (x:64 + folded agg:256), WMMA ----
  {
    const int t = wave >> 1, nh = wave & 1;
    v8f c = {};
#pragma unroll
    for (int sstep = 0; sstep < 10; ++sstep) {
      int ka = (sstep < 2) ? sstep * 32 : 64 + t * 256 + (sstep - 2) * 32;
      v16h a = load_a16(A, ASTR, ka, lane);
      v16h b = load_b16(Weff2T + t * 32 * 320, 320, nh * 16, sstep * 32, lane);
      c = wmma_f16(a, b, c);
    }
    int colL = nh * 16 + (lane & 15);  // 0..31 within tower
    int col = t * 32 + colL;           // 0..127
    float bb = bpost2[t * 32 + colL];
#pragma unroll
    for (int v = 0; v < 8; ++v) {
      int row = v + ((lane >> 4) * 8);
      A2[row * 130 + col] = (_Float16)(c[v] + bb);
    }
  }
  __syncthreads();
  // ---- phase 3: mixing linear [16,128]x[128,128], WMMA ----
  {
    const int n0 = wave * 16;
    v8f c = {};
#pragma unroll
    for (int sstep = 0; sstep < 4; ++sstep) {
      v16h a = load_a16(A2, 130, sstep * 32, lane);
      v16h b = load_b16(Wlin2T, 128, n0, sstep * 32, lane);
      c = wmma_f16(a, b, c);
    }
    int col = n0 + (lane & 15);
    float bb = blin2[col];
#pragma unroll
    for (int v = 0; v < 8; ++v) {
      int row = v + ((lane >> 4) * 8);
      H2[(size_t)(m0 + row) * 128 + col] = c[v] + bb;
    }
  }
}

// ---------------- final: BN + ReLU + per-graph mean pool ----------------
__global__ void k_final(const float* __restrict__ H2, const float* __restrict__ mean,
                        const float* __restrict__ istd, const float* __restrict__ g,
                        const float* __restrict__ b, float* __restrict__ out) {
  const int gr = blockIdx.x, c = threadIdx.x;  // 128 threads = channels
  float m = mean[c], is = istd[c], gg = g[c], bb = b[c];
  float s = 0.f;
  for (int i = 0; i < NPG; ++i) {
    float v = gg * (H2[(size_t)(gr * NPG + i) * 128 + c] - m) * is + bb;
    s += fmaxf(v, 0.f);
  }
  out[gr * 128 + c] = s * (1.0f / (float)NPG);
}

// ---------------- host launcher ----------------
static inline char* carve(char*& p, size_t bytes) {
  char* r = p;
  p += (bytes + 255) & ~(size_t)255;
  return r;
}

extern "C" void kernel_launch(void* const* d_in, const int* in_sizes, int n_in,
                              void* d_out, int out_size, void* d_ws, size_t ws_size,
                              hipStream_t stream) {
  const float* X       = (const float*)d_in[0];
  const float* pos     = (const float*)d_in[1];
  const float* W_pre1  = (const float*)d_in[2];
  const float* b_pre1  = (const float*)d_in[3];
  const float* W_post1 = (const float*)d_in[4];
  const float* b_post1 = (const float*)d_in[5];
  const float* W_lin1  = (const float*)d_in[6];
  const float* b_lin1  = (const float*)d_in[7];
  const float* bn1_g   = (const float*)d_in[8];
  const float* bn1_b   = (const float*)d_in[9];
  const float* W_pre2  = (const float*)d_in[10];
  const float* b_pre2  = (const float*)d_in[11];
  const float* W_post2 = (const float*)d_in[12];
  const float* b_post2 = (const float*)d_in[13];
  const float* W_lin2  = (const float*)d_in[14];
  const float* b_lin2  = (const float*)d_in[15];
  const float* bn2_g   = (const float*)d_in[16];
  const float* bn2_b   = (const float*)d_in[17];
  float* out = (float*)d_out;

  char* p = (char*)d_ws;
  int*       nbr    = (int*)      carve(p, (size_t)NTOT * KNN * 4);
  float*     P1     = (float*)    carve(p, (size_t)NTOT * 64 * 4);
  float*     H1     = (float*)    carve(p, (size_t)NTOT * 64 * 4);
  _Float16*  X2h    = (_Float16*) carve(p, (size_t)NTOT * 64 * 2);
  float*     PA2    = (float*)    carve(p, (size_t)NTOT * 256 * 4);
  float*     PB2    = (float*)    carve(p, (size_t)NTOT * 256 * 4);
  float*     H2     = (float*)    carve(p, (size_t)NTOT * 128 * 4);
  float*     bn1m   = (float*)    carve(p, 64 * 4);
  float*     bn1i   = (float*)    carve(p, 64 * 4);
  float*     bn2m   = (float*)    carve(p, 128 * 4);
  float*     bn2i   = (float*)    carve(p, 128 * 4);
  float*     Wcat1  = (float*)    carve(p, 512 * 4);
  float*     bpre1c = (float*)    carve(p, 32 * 4);
  float*     Weff1  = (float*)    carve(p, 2560 * 4);
  float*     bpre2c = (float*)    carve(p, 256 * 4);
  _Float16*  WcatT  = (_Float16*) carve(p, 512 * 64 * 2);
  _Float16*  Weff2T = (_Float16*) carve(p, 4 * 32 * 320 * 2);
  _Float16*  Wlin2T = (_Float16*) carve(p, 128 * 128 * 2);

  k_prep<<<1, 256, 0, stream>>>(W_pre1, b_pre1, W_post1, W_pre2, b_pre2, W_post2,
                                W_lin2, Wcat1, bpre1c, Weff1, bpre2c,
                                WcatT, Weff2T, Wlin2T);
  k_knn<<<NG, 256, 0, stream>>>(pos, nbr);
  k_l1_pre<<<NTOT / 4, 256, 0, stream>>>(X, Wcat1, bpre1c, P1);
  k_l1_agg_post<<<NTOT / 32, 128, 0, stream>>>(X, P1, nbr, Weff1, W_lin1,
                                               b_post1, b_lin1, H1);
  k_bnstats<<<64, 256, 0, stream>>>(H1, 64, NTOT, bn1m, bn1i);
  k_bn_relu_half<<<(NTOT * 64) / 256, 256, 0, stream>>>(H1, bn1m, bn1i, bn1_g, bn1_b,
                                                        X2h, 64, NTOT * 64);
  k2_pre<<<NTOT / 16, 256, 0, stream>>>(X2h, WcatT, bpre2c, PA2, PB2);
  k2_agg_post<<<NTOT / 16, 256, 0, stream>>>(PA2, PB2, X2h, nbr, Weff2T, Wlin2T,
                                             b_post2, b_lin2, H2);
  k_bnstats<<<128, 256, 0, stream>>>(H2, 128, NTOT, bn2m, bn2i);
  k_final<<<NG, 128, 0, stream>>>(H2, bn2m, bn2i, bn2_g, bn2_b, out);
  (void)in_sizes; (void)n_in; (void)out_size; (void)ws_size;
}
